// PlotLine2_52261162058496
// MI455X (gfx1250) — matrix operand order repaired and checked by reading, not compiled, not run on hardware
//
#include <hip/hip_runtime.h>

typedef __attribute__((ext_vector_type(16))) _Float16 v16h;
typedef __attribute__((ext_vector_type(8)))  _Float16 v8h;
typedef __attribute__((ext_vector_type(8)))  float    v8f;

#define NBATCH 128
#define NPTS   16
#define SDIM   128
#define KTOT   1920   // (NPTS-1)*SDIM
#define KC     32     // K per chunk (WMMA f16 K)
#define STRIP  64     // output columns per block

#define NEG_LOG2E (-1.44269504088896340736f)

__device__ __forceinline__ unsigned pack_gauss(float d0, float d1) {
  // exp(-d^2) via raw v_exp_f32; d is an integer so only ~11 nonzero f16 levels exist
  float e0 = __builtin_amdgcn_exp2f(d0 * d0 * NEG_LOG2E);
  float e1 = __builtin_amdgcn_exp2f(d1 * d1 * NEG_LOG2E);
  union { _Float16 h[2]; unsigned u; } pk;
  pk.h[0] = (_Float16)e0;
  pk.h[1] = (_Float16)e1;
  return pk.u;
}

__global__ __launch_bounds__(256)
void plotline_wmma_kernel(const float* __restrict__ points,
                          float* __restrict__ out) {
  // position-major, k-minor so fragments are contiguous 16B LDS loads
  __shared__ __align__(16) _Float16 sgx[SDIM][KC];   // 8 KB
  __shared__ __align__(16) _Float16 sgy[STRIP][KC];  // 4 KB

  const int n     = blockIdx.x;       // batch
  const int strip = blockIdx.y;       // 0..1 (column strip of 64)
  const int tid   = threadIdx.x;
  const int wid   = tid >> 5;         // 8 waves
  const int lane  = tid & 31;

  const int wave_m = wid >> 2;        // 0..1  -> rows [wave_m*64, +64)
  const int wave_n = wid & 3;         // 0..3  -> cols [wave_n*16, +16) in strip
  const int t0     = strip * STRIP + wave_n * 16;

  const float* pbase = points + n * NPTS * 2;

  v8f acc[4] = {v8f{}, v8f{}, v8f{}, v8f{}};

  for (int kc = 0; kc < KTOT; kc += KC) {
    __syncthreads();   // previous chunk's fragment loads complete
    {
      // each fill thread owns a kk *pair* -> packed b32 LDS stores
      const int c   = tid & 15;              // pair index: kk = 2c, 2c+1
      const int k0  = kc + 2 * c;
      const int seg0 = k0 >> 7;
      const int i0   = k0 & 127;
      const int k1  = k0 + 1;
      const int seg1 = k1 >> 7;
      const int i1   = k1 & 127;
      const float tt0 = (float)i0 * (1.0f / 128.0f);
      const float tt1 = (float)i1 * (1.0f / 128.0f);
      if (tid < 128) {
        const float x0 = __builtin_rintf((1.0f - tt0) * pbase[seg0 * 2 + 0] +
                                         tt0 * pbase[(seg0 + 1) * 2 + 0]);
        const float x1 = __builtin_rintf((1.0f - tt1) * pbase[seg1 * 2 + 0] +
                                         tt1 * pbase[(seg1 + 1) * 2 + 0]);
        const int sb = (tid >> 4) * 16;      // 8 groups x 16 rows = 128 s
        #pragma unroll
        for (int ss = 0; ss < 16; ++ss) {
          const float sv = (float)(sb + ss);
          *(unsigned*)&sgx[sb + ss][2 * c] = pack_gauss(x0 - sv, x1 - sv);
        }
      } else {
        const float y0 = __builtin_rintf((1.0f - tt0) * pbase[seg0 * 2 + 1] +
                                         tt0 * pbase[(seg0 + 1) * 2 + 1]);
        const float y1 = __builtin_rintf((1.0f - tt1) * pbase[seg1 * 2 + 1] +
                                         tt1 * pbase[(seg1 + 1) * 2 + 1]);
        const int tb = ((tid - 128) >> 4) * 8;  // 8 groups x 8 rows = 64 t
        #pragma unroll
        for (int ttl = 0; ttl < 8; ++ttl) {
          const float tv = (float)(strip * STRIP + tb + ttl);
          *(unsigned*)&sgy[tb + ttl][2 * c] = pack_gauss(y0 - tv, y1 - tv);
        }
      }
    }
    __syncthreads();

    // B fragment (32x16): lane<16 -> N=lane, K 0-7 & 16-23 ; lane>=16 -> K 8-15 & 24-31
    const int tl    = wave_n * 16 + (lane & 15);
    const int bbase = (lane < 16) ? 0 : 8;
    v16h bf;
    *(v8h*)&bf       = *(const v8h*)&sgy[tl][bbase];
    *((v8h*)&bf + 1) = *(const v8h*)&sgy[tl][bbase + 16];

    // preload all 4 A fragments, then burst the WMMAs
    v16h af[4];
    #pragma unroll
    for (int a = 0; a < 4; ++a) {
      // A fragment (16x32): lane<16 -> M=lane, K 0-7 & 16-23 ; lane>=16 -> K 8-15 & 24-31
      const int sl = (wave_m * 4 + a) * 16 + (lane & 15);
      const int ab = (lane < 16) ? 0 : 8;
      *(v8h*)&af[a]       = *(const v8h*)&sgx[sl][ab];
      *((v8h*)&af[a] + 1) = *(const v8h*)&sgx[sl][ab + 16];
    }
    #pragma unroll
    for (int a = 0; a < 4; ++a) {
      acc[a] = __builtin_amdgcn_wmma_f32_16x16x32_f16(
          /*neg_a=*/false, af[a], /*neg_b=*/false, bf,
          /*c_mod=*/(short)0, acc[a], /*reuse_a=*/false, /*reuse_b=*/false);
    }
  }

  // Epilogue: tanh + store. C/D layout: lane%16 = N col; VGPR j = row j (lanes<16) / j+8
  const int tg = t0 + (lane & 15);
  float* obase = out + (size_t)n * SDIM * SDIM;
  #pragma unroll
  for (int a = 0; a < 4; ++a) {
    const int s0 = (wave_m * 4 + a) * 16;
    #pragma unroll
    for (int j = 0; j < 8; ++j) {
      const int m = s0 + ((lane < 16) ? j : j + 8);
      obase[m * SDIM + tg] = tanhf(acc[a][j]);
    }
  }
}

extern "C" void kernel_launch(void* const* d_in, const int* in_sizes, int n_in,
                              void* d_out, int out_size, void* d_ws, size_t ws_size,
                              hipStream_t stream) {
  (void)in_sizes; (void)n_in; (void)d_ws; (void)ws_size; (void)out_size;
  const float* points = (const float*)d_in[0];   // [128,16,2] f32
  float* out = (float*)d_out;                    // [128,128,128] f32
  dim3 grid(NBATCH, SDIM / STRIP, 1);            // 128 x 2 blocks
  dim3 block(256, 1, 1);                         // 8 wave32s
  plotline_wmma_kernel<<<grid, block, 0, stream>>>(points, out);
}
